// MoE_56066503082696
// MI455X (gfx1250) — compile-verified, matrix operand order checked by reading, not run on hardware
//
#include <hip/hip_runtime.h>
#include <hip/hip_bf16.h>

#define DIM     2048
#define HIDDEN  1408
#define NEXP    32
#define TOPK    4
#define NTOK    4096
#define NK      (NTOK * TOPK)

#define BM      32      // routed rows per block
#define BN      64      // output cols per tile
#define KC      32      // K step (bf16 WMMA K)
#define APITCH  40      // LDS pitch (bf16 elems) for A/B staging tiles
#define HPITCH  1416    // LDS pitch (bf16 elems) for h tile

typedef __attribute__((ext_vector_type(16))) __bf16 v16bf;
typedef __attribute__((ext_vector_type(4)))  __bf16 v4bf;
typedef __attribute__((ext_vector_type(8)))  float  v8f;

union FragU { struct { uint4 lo; uint4 hi; } q; v16bf v; };

// ---- f32 -> bf16 (RNE) via native clang conversion: lets the gfx1250
// backend select the hardware cvt instruction(s) instead of bit math. ----
__device__ __forceinline__ __bf16 f32_to_bf16(float f) {
    return (__bf16)f;
}

// stage 4 consecutive f32 -> 4 bf16 (8 bytes) into LDS
__device__ __forceinline__ void stage4(__bf16* dst, const float* src) {
    const float4 v = *reinterpret_cast<const float4*>(src);
    v4bf p;
    p[0] = (__bf16)v.x; p[1] = (__bf16)v.y;
    p[2] = (__bf16)v.z; p[3] = (__bf16)v.w;
    *reinterpret_cast<v4bf*>(dst) = p;   // 8-byte LDS store
}

// Load one 16x32 (MxK) bf16 fragment from a row-major LDS tile.
// Per CDNA5 ISA 7.12.2: lane l (l<16) holds row base+l, K in {0..7, 16..23};
// lane l+16 holds K in {8..15, 24..31} -> two contiguous 16B chunks per lane.
__device__ __forceinline__ v16bf load_frag_lds(const __bf16* base, int pitch,
                                               int rowBase, int kBase, int lane) {
    int half = lane >> 4;
    int r    = lane & 15;
    const __bf16* p = base + (size_t)(rowBase + r) * pitch + kBase + half * 8;
    FragU u;
    u.q.lo = *reinterpret_cast<const uint4*>(p);
    u.q.hi = *reinterpret_cast<const uint4*>(p + 16);
    return u.v;
}

// ---------------- routing kernels (deterministic) ----------------

__global__ void moe_hist_kernel(const int* __restrict__ idx, int* __restrict__ cnt) {
    int i = blockIdx.x * blockDim.x + threadIdx.x;
    if (i < NK) atomicAdd(&cnt[idx[i]], 1);
}

__global__ void moe_scan_kernel(const int* __restrict__ cnt,
                                int* __restrict__ offs, int* __restrict__ tp) {
    if (blockIdx.x == 0 && threadIdx.x == 0) {
        int o = 0, t = 0;
        offs[0] = 0; tp[0] = 0;
        for (int e = 0; e < NEXP; ++e) {
            o += cnt[e];                     offs[e + 1] = o;
            t += (cnt[e] + BM - 1) / BM;     tp[e + 1]   = t;
        }
    }
}

// Stable compaction: pos(i) = offs[e_i] + |{j < i : e_j == e_i}| (matches
// jnp.argsort stable order). O(NK) scan per thread, cache-resident.
__global__ void moe_rank_kernel(const int* __restrict__ idx,
                                const float* __restrict__ scores,
                                const int* __restrict__ offs,
                                int* __restrict__ rTok, float* __restrict__ rScore) {
    int i = blockIdx.x * blockDim.x + threadIdx.x;
    if (i >= NK) return;
    int e = idx[i];
    int r = 0;
    for (int j = 0; j < i; ++j) r += (idx[j] == e) ? 1 : 0;
    int pos = offs[e] + r;
    rTok[pos]   = i / TOPK;
    rScore[pos] = scores[i];
}

// ---------------- fused expert kernel ----------------
// One block = (expert, 32-row block). 128 threads = 4 wave32s.
// Wave w: row group rg = w>>1 (16 rows), col group cg = w&1 (32 cols of BN).
// Phase 1: h = silu(x W1^T) * (x W3^T) -> bf16 tile in LDS (32 x 1408).
// Phase 2: out_r = h W2^T, scaled by router score, atomic-add to out.

__global__ void __launch_bounds__(128, 1)
moe_fused_kernel(const float* __restrict__ x,
                 const float* __restrict__ w1,
                 const float* __restrict__ w2,
                 const float* __restrict__ w3,
                 const int* __restrict__ cnt,
                 const int* __restrict__ offs,
                 const int* __restrict__ tp,
                 const int* __restrict__ rTok,
                 const float* __restrict__ rScore,
                 float* __restrict__ out) {
    __shared__ __bf16 sH[BM][HPITCH];          // 90,624 B : full h tile
    __shared__ __bf16 sA[BM][APITCH];          //  2,560 B : activations
    __shared__ __bf16 sB[2][BN][APITCH];       // 10,240 B : weights
    __shared__ int    sTok[BM];
    __shared__ float  sScore[BM];
    __shared__ int    sMeta[4];

    const int b   = blockIdx.x;
    const int tid = threadIdx.x;

    if (tid == 0) {
        if (b >= tp[NEXP]) {
            sMeta[0] = -1;
        } else {
            int e = 0;
            while (tp[e + 1] <= b) ++e;          // monotone prefix search
            int rb = b - tp[e];
            sMeta[0] = e;
            sMeta[1] = offs[e] + rb * BM;
            int v = cnt[e] - rb * BM;
            sMeta[2] = v < BM ? v : BM;
        }
    }
    __syncthreads();
    if (sMeta[0] < 0) return;                    // uniform exit
    const int e = sMeta[0], rowStart = sMeta[1], vRows = sMeta[2];

    if (tid < BM) {
        if (tid < vRows) { sTok[tid] = rTok[rowStart + tid]; sScore[tid] = rScore[rowStart + tid]; }
        else             { sTok[tid] = -1;                   sScore[tid] = 0.f; }
    }
    __syncthreads();

    const int lane = tid & 31;
    const int w    = tid >> 5;
    const int rg   = w >> 1;
    const int cg   = w & 1;

    const float* w1e = w1 + (size_t)e * HIDDEN * DIM;
    const float* w3e = w3 + (size_t)e * HIDDEN * DIM;
    const float* w2e = w2 + (size_t)e * DIM * HIDDEN;

    // ---- hoisted per-thread staging coordinates (vec4 granularity) ----
    // A tile: 32 rows x 8 vec4s = 256 chunks, 2 per thread
    const int ar0 = tid >> 3;            // 0..15
    const int akv = (tid & 7) * 4;       // element offset within row
    int t0 = sTok[ar0];      t0 = t0 < 0 ? 0 : t0;
    int t1 = sTok[ar0 + 16]; t1 = t1 < 0 ? 0 : t1;
    const float* xp0 = x + (size_t)t0 * DIM + akv;
    const float* xp1 = x + (size_t)t1 * DIM + akv;
    __bf16* sA0 = &sA[ar0][akv];
    __bf16* sA1 = &sA[ar0 + 16][akv];
    // B tile: 64 rows x 8 vec4s = 512 chunks, 4 per thread (rows br0+16p)
    const int br0 = tid >> 3;            // 0..15

    // -------- Phase 1: h = silu(x W1^T) * (x W3^T) --------
    for (int ct = 0; ct < HIDDEN / BN; ++ct) {
        v8f acc1a = {}, acc1b = {}, acc3a = {}, acc3b = {};
        const float* b1p = w1e + (size_t)(ct * BN + br0) * DIM + akv;
        const float* b3p = w3e + (size_t)(ct * BN + br0) * DIM + akv;
        for (int kk = 0; kk < DIM; kk += KC) {
            // stage A: BM x KC activations (gathered rows), f32 -> bf16
            stage4(sA0, xp0 + kk);
            stage4(sA1, xp1 + kk);
            // stage B: BN x KC rows of w1 and w3
            #pragma unroll
            for (int p = 0; p < 4; ++p) {
                stage4(&sB[0][br0 + 16 * p][akv], b1p + (size_t)16 * p * DIM + kk);
                stage4(&sB[1][br0 + 16 * p][akv], b3p + (size_t)16 * p * DIM + kk);
                if (kk + KC < DIM) {
                    __builtin_prefetch(b1p + (size_t)16 * p * DIM + kk + KC, 0, 1);
                    __builtin_prefetch(b3p + (size_t)16 * p * DIM + kk + KC, 0, 1);
                }
            }
            __syncthreads();

            v16bf af  = load_frag_lds(&sA[0][0],    APITCH, rg * 16,      0, lane);
            v16bf b1a = load_frag_lds(&sB[0][0][0], APITCH, cg * 32,      0, lane);
            v16bf b1b = load_frag_lds(&sB[0][0][0], APITCH, cg * 32 + 16, 0, lane);
            v16bf b3a = load_frag_lds(&sB[1][0][0], APITCH, cg * 32,      0, lane);
            v16bf b3b = load_frag_lds(&sB[1][0][0], APITCH, cg * 32 + 16, 0, lane);

            acc1a = __builtin_amdgcn_wmma_f32_16x16x32_bf16(false, af, false, b1a, (short)0, acc1a, false, false);
            acc1b = __builtin_amdgcn_wmma_f32_16x16x32_bf16(false, af, false, b1b, (short)0, acc1b, false, false);
            acc3a = __builtin_amdgcn_wmma_f32_16x16x32_bf16(false, af, false, b3a, (short)0, acc3a, false, false);
            acc3b = __builtin_amdgcn_wmma_f32_16x16x32_bf16(false, af, false, b3b, (short)0, acc3b, false, false);
            __syncthreads();
        }
        // epilogue: SwiGLU, write bf16 into LDS h tile.
        // C layout (ISA 7.12.2): VGPR i -> M=i (lanes 0-15) / M=8+i (lanes 16-31), N=lane&15
        int n0    = ct * BN + cg * 32 + (lane & 15);
        int rbase = rg * 16 + ((lane < 16) ? 0 : 8);
        #pragma unroll
        for (int i = 0; i < 8; ++i) {
            float a1 = acc1a[i];
            float h1 = (a1 / (1.f + __expf(-a1))) * acc3a[i];
            sH[rbase + i][n0] = f32_to_bf16(h1);
            float a2 = acc1b[i];
            float h2 = (a2 / (1.f + __expf(-a2))) * acc3b[i];
            sH[rbase + i][n0 + 16] = f32_to_bf16(h2);
        }
    }
    __syncthreads();

    // -------- Phase 2: out_r = h W2^T, scatter-add weighted by score --------
    for (int ct = 0; ct < DIM / BN; ++ct) {
        v8f acc0 = {}, acc1 = {};
        const float* b2p = w2e + (size_t)(ct * BN + br0) * HIDDEN + akv;
        for (int kk = 0; kk < HIDDEN; kk += KC) {
            #pragma unroll
            for (int p = 0; p < 4; ++p) {
                stage4(&sB[0][br0 + 16 * p][akv], b2p + (size_t)16 * p * HIDDEN + kk);
                if (kk + KC < HIDDEN)
                    __builtin_prefetch(b2p + (size_t)16 * p * HIDDEN + kk + KC, 0, 1);
            }
            __syncthreads();

            v16bf af = load_frag_lds(&sH[0][0],    HPITCH, rg * 16,      kk, lane);
            v16bf b0 = load_frag_lds(&sB[0][0][0], APITCH, cg * 32,      0,  lane);
            v16bf b1 = load_frag_lds(&sB[0][0][0], APITCH, cg * 32 + 16, 0,  lane);

            acc0 = __builtin_amdgcn_wmma_f32_16x16x32_bf16(false, af, false, b0, (short)0, acc0, false, false);
            acc1 = __builtin_amdgcn_wmma_f32_16x16x32_bf16(false, af, false, b1, (short)0, acc1, false, false);
            __syncthreads();
        }
        int n0    = ct * BN + cg * 32 + (lane & 15);
        int rbase = rg * 16 + ((lane < 16) ? 0 : 8);
        #pragma unroll
        for (int i = 0; i < 8; ++i) {
            int rl = rbase + i;
            int t  = sTok[rl];
            if (t >= 0) {
                float s = sScore[rl];
                atomicAdd(&out[(size_t)t * DIM + n0],      acc0[i] * s);
                atomicAdd(&out[(size_t)t * DIM + n0 + 16], acc1[i] * s);
            }
        }
    }
}

// ---------------- host launcher ----------------

extern "C" void kernel_launch(void* const* d_in, const int* in_sizes, int n_in,
                              void* d_out, int out_size, void* d_ws, size_t ws_size,
                              hipStream_t stream) {
    const float* x      = (const float*)d_in[0];
    const float* scores = (const float*)d_in[1];
    const int*   idx    = (const int*)  d_in[2];   // jax default int (x64 disabled)
    const float* w1     = (const float*)d_in[3];
    const float* w2     = (const float*)d_in[4];
    const float* w3     = (const float*)d_in[5];
    float* out = (float*)d_out;

    // workspace layout (ints): cnt[32] | offs[33] | tp[33] | pad to 128 | rTok[NK] | rScore[NK]
    int*   cnt    = (int*)d_ws;
    int*   offs   = cnt + 32;
    int*   tp     = offs + 33;
    int*   rTok   = cnt + 128;
    float* rScore = (float*)(rTok + NK);

    hipMemsetAsync(out, 0, (size_t)NTOK * DIM * sizeof(float), stream);
    hipMemsetAsync(cnt, 0, 32 * sizeof(int), stream);

    moe_hist_kernel<<<NK / 256, 256, 0, stream>>>(idx, cnt);
    moe_scan_kernel<<<1, 32, 0, stream>>>(cnt, offs, tp);
    moe_rank_kernel<<<NK / 256, 256, 0, stream>>>(idx, scores, offs, rTok, rScore);

    // worst-case tiles: sum_e ceil(cnt_e/BM) <= NK/BM + NEXP
    const int maxTiles = NK / BM + NEXP;   // 544
    moe_fused_kernel<<<maxTiles, 128, 0, stream>>>(x, w1, w2, w3,
                                                   cnt, offs, tp, rTok, rScore, out);
}
